// Seq2Seq_71966472012477
// MI455X (gfx1250) — matrix operand order repaired and checked by reading, not compiled, not run on hardware
//
#include <hip/hip_runtime.h>

// ---------------------------------------------------------------------------
// Seq2Seq (biLSTM encoder + attention + parallel decoder) for gfx1250.
// All GEMMs use V_WMMA_F32_16X16X32_BF16 (bf16 in, fp32 accumulate).
// ---------------------------------------------------------------------------

#define SLEN   64
#define BATCH  512
#define ES     256
#define HS     512
#define NTOK   47          // TRG_LEN - 1
#define OUTD   67

typedef __attribute__((ext_vector_type(16))) __bf16 v16bf;
typedef __attribute__((ext_vector_type(8)))  float  v8f;

__device__ __forceinline__ float sigm(float x) { return 1.f / (1.f + __expf(-x)); }

// A fragment (16x32 bf16, MxK): lane half h: a[0..7] = K(kb + h*8 + i),
// a[8..15] = K(kb + 16 + h*8 + i)  -> two contiguous 16B loads.
__device__ __forceinline__ v16bf ld_fragA(const __bf16* rowp, int kb, int half) {
  union { v16bf v; uint4 u[2]; } r;
  const __bf16* p = rowp + kb + half * 8;
  r.u[0] = *(const uint4*)(p);
  r.u[1] = *(const uint4*)(p + 16);
  return r.v;
}

// B fragment (32x16 bf16, KxN, fed from row-major W[n][k] = B^T): lane half h:
// b[0..7] = K(kb + h*8 + i), b[8..15] = K(kb + 16 + h*8 + i) -> two 16B loads.
__device__ __forceinline__ v16bf ld_fragB(const __bf16* colp, int kb, int half) {
  union { v16bf v; uint4 u[2]; } r;
  const __bf16* p = colp + kb + half * 8;
  r.u[0] = *(const uint4*)(p);
  r.u[1] = *(const uint4*)(p + 16);
  return r.v;
}

__device__ __forceinline__ v8f wmma_bf(v16bf a, v16bf b, v8f c) {
  return __builtin_amdgcn_wmma_f32_16x16x32_bf16(false, a, false, b, (short)0, c,
                                                 false, false);
}

__device__ __forceinline__ void mma_2x2(const __bf16* a0, const __bf16* a1,
                                        const __bf16* b0, const __bf16* b1,
                                        int K, int half, v8f (&acc)[2][2]) {
  for (int kb = 0; kb < K; kb += 32) {
    v16bf fa0 = ld_fragA(a0, kb, half);
    v16bf fa1 = ld_fragA(a1, kb, half);
    v16bf fb0 = ld_fragB(b0, kb, half);
    v16bf fb1 = ld_fragB(b1, kb, half);
    acc[0][0] = wmma_bf(fa0, fb0, acc[0][0]);
    acc[0][1] = wmma_bf(fa0, fb1, acc[0][1]);
    acc[1][0] = wmma_bf(fa1, fb0, acc[1][0]);
    acc[1][1] = wmma_bf(fa1, fb1, acc[1][1]);
  }
}

// ---------------------------------------------------------------------------
// small utility kernels
// ---------------------------------------------------------------------------
__global__ void cvt_kernel(const float* __restrict__ in, __bf16* __restrict__ o, int n) {
  int i = blockIdx.x * blockDim.x + threadIdx.x;
  if (i < n) o[i] = (__bf16)in[i];
}

__global__ void zero_kernel(float* p, int n) {
  int i = blockIdx.x * blockDim.x + threadIdx.x;
  if (i < n) p[i] = 0.f;
}

__global__ void gather_kernel(const int* __restrict__ tok, const float* __restrict__ tab,
                              __bf16* __restrict__ out, int rows) {
  int i = blockIdx.x * blockDim.x + threadIdx.x;
  if (i >= rows * ES) return;
  int r = i >> 8;          // ES == 256
  int e = i & 255;
  out[i] = (__bf16)tab[tok[r] * ES + e];
}

// ---------------------------------------------------------------------------
// encoder step: fused (X@Wih^T + H@Whh^T) WMMA GEMM + LSTM pointwise.
// grid (B/64, HS/32, 2 dirs), 256 threads = 8 waves = 2(M) x 4(gate).
// ---------------------------------------------------------------------------
struct EncParams {
  const __bf16* emb;                       // [S][B][ES]
  const __bf16* wih0; const __bf16* whh0;  // fwd  (4HS x ES), (4HS x HS)
  const __bf16* wih1; const __bf16* whh1;  // bwd
  const float*  bih0; const float*  bhh0;
  const float*  bih1; const float*  bhh1;
  const __bf16* hrd0; const __bf16* hrd1;  // h read  (bf16, double buffered)
  __bf16*       hwr0; __bf16*       hwr1;  // h write
  float* c0; float* c1;                    // cell state fp32 [B][HS]
  float* out0; float* out1;                // [S][B][HS]
  int t;
};

__global__ __launch_bounds__(256) void enc_step_kernel(EncParams p) {
  const int dir = blockIdx.z;
  const int t   = p.t;
  const int s   = dir ? (SLEN - 1 - t) : t;
  const __bf16* wih = dir ? p.wih1 : p.wih0;
  const __bf16* whh = dir ? p.whh1 : p.whh0;
  const float*  bih = dir ? p.bih1 : p.bih0;
  const float*  bhh = dir ? p.bhh1 : p.bhh0;
  const __bf16* hrd = dir ? p.hrd1 : p.hrd0;
  __bf16*       hwr = dir ? p.hwr1 : p.hwr0;
  float*        cst = dir ? p.c1   : p.c0;
  float*        out = dir ? p.out1 : p.out0;

  const int tid  = threadIdx.x;
  const int wave = tid >> 5, lane = tid & 31;
  const int waveM = wave >> 2, gate = wave & 3;
  const int half = lane >> 4, l16 = lane & 15;
  const int rowBase = blockIdx.x * 64 + waveM * 32;
  const int colGate = blockIdx.y * 32;

  const v8f vz = {0.f, 0.f, 0.f, 0.f, 0.f, 0.f, 0.f, 0.f};
  v8f acc[2][2] = {{vz, vz}, {vz, vz}};

  {  // x_t @ Wih^T, K = ES
    const __bf16* X  = p.emb + (size_t)s * BATCH * ES;
    const __bf16* a0 = X + (size_t)(rowBase + l16) * ES;
    const __bf16* a1 = X + (size_t)(rowBase + 16 + l16) * ES;
    const int j0 = gate * HS + colGate + l16;
    const __bf16* b0 = wih + (size_t)j0 * ES;
    const __bf16* b1 = wih + (size_t)(j0 + 16) * ES;
    mma_2x2(a0, a1, b0, b1, ES, half, acc);
  }
  if (t > 0) {  // h @ Whh^T, K = HS (h == 0 at t == 0)
    const __bf16* a0 = hrd + (size_t)(rowBase + l16) * HS;
    const __bf16* a1 = hrd + (size_t)(rowBase + 16 + l16) * HS;
    const int j0 = gate * HS + colGate + l16;
    const __bf16* b0 = whh + (size_t)j0 * HS;
    const __bf16* b1 = whh + (size_t)(j0 + 16) * HS;
    mma_2x2(a0, a1, b0, b1, HS, half, acc);
  }

  __shared__ float g_lds[4][64][33];
  for (int mt = 0; mt < 2; ++mt)
    for (int nt = 0; nt < 2; ++nt)
      for (int r = 0; r < 8; ++r)
        g_lds[gate][waveM * 32 + mt * 16 + half * 8 + r][nt * 16 + l16] = acc[mt][nt][r];
  __syncthreads();

  const int col = tid & 31;
  const int rg  = tid >> 5;
  const int n   = blockIdx.y * 32 + col;
  const float bi  = bih[n]          + bhh[n];
  const float bff = bih[HS + n]     + bhh[HS + n];
  const float bgg = bih[2 * HS + n] + bhh[2 * HS + n];
  const float boo = bih[3 * HS + n] + bhh[3 * HS + n];
  for (int r = 0; r < 8; ++r) {
    const int rl  = rg * 8 + r;
    const int row = blockIdx.x * 64 + rl;
    float gi = g_lds[0][rl][col] + bi;
    float gf = g_lds[1][rl][col] + bff;
    float gg = g_lds[2][rl][col] + bgg;
    float go = g_lds[3][rl][col] + boo;
    float cold = (t == 0) ? 0.f : cst[(size_t)row * HS + n];
    float cn = sigm(gf) * cold + sigm(gi) * tanhf(gg);
    float hn = sigm(go) * tanhf(cn);
    cst[(size_t)row * HS + n] = cn;
    out[((size_t)s * BATCH + row) * HS + n] = hn;
    hwr[(size_t)row * HS + n] = (__bf16)hn;
  }
}

// ---------------------------------------------------------------------------
// generic two-K-segment WMMA GEMM:  C = [A0|A1] @ [B0|B1]^T + bias0 (+bias1)
// grid (M/64, N/128), 256 threads = 8 waves = 2(M) x 4(N).
// ---------------------------------------------------------------------------
struct GemmParams {
  const __bf16* A0; int lda0; int K0;
  const __bf16* A1; int lda1; int K1;
  const __bf16* B0; int ldb0;         // row-major (N x K0)
  const __bf16* B1; int ldb1;         // row-major (N x K1)
  const float* bias0; const float* bias1;
  float* Cf; __bf16* Cb; int ldc;
};

__global__ __launch_bounds__(256) void gemm2seg_kernel(GemmParams p) {
  const int tid  = threadIdx.x;
  const int wave = tid >> 5, lane = tid & 31;
  const int waveM = wave >> 2, waveN = wave & 3;
  const int half = lane >> 4, l16 = lane & 15;
  const int rowBase = blockIdx.x * 64 + waveM * 32;
  const int colBase = blockIdx.y * 128 + waveN * 32;

  const v8f vz = {0.f, 0.f, 0.f, 0.f, 0.f, 0.f, 0.f, 0.f};
  v8f acc[2][2] = {{vz, vz}, {vz, vz}};

  {
    const __bf16* a0 = p.A0 + (size_t)(rowBase + l16) * p.lda0;
    const __bf16* a1 = p.A0 + (size_t)(rowBase + 16 + l16) * p.lda0;
    const __bf16* b0 = p.B0 + (size_t)(colBase + l16) * p.ldb0;
    const __bf16* b1 = p.B0 + (size_t)(colBase + 16 + l16) * p.ldb0;
    mma_2x2(a0, a1, b0, b1, p.K0, half, acc);
  }
  if (p.K1 > 0) {
    const __bf16* a0 = p.A1 + (size_t)(rowBase + l16) * p.lda1;
    const __bf16* a1 = p.A1 + (size_t)(rowBase + 16 + l16) * p.lda1;
    const __bf16* b0 = p.B1 + (size_t)(colBase + l16) * p.ldb1;
    const __bf16* b1 = p.B1 + (size_t)(colBase + 16 + l16) * p.ldb1;
    mma_2x2(a0, a1, b0, b1, p.K1, half, acc);
  }

  for (int mt = 0; mt < 2; ++mt)
    for (int nt = 0; nt < 2; ++nt)
      for (int r = 0; r < 8; ++r) {
        const int row = rowBase + mt * 16 + half * 8 + r;
        const int col = colBase + nt * 16 + l16;
        float v = acc[mt][nt][r];
        if (p.bias0) v += p.bias0[col];
        if (p.bias1) v += p.bias1[col];
        if (p.Cf) p.Cf[(size_t)row * p.ldc + col] = v;
        if (p.Cb) p.Cb[(size_t)row * p.ldc + col] = (__bf16)v;
      }
}

// ---------------------------------------------------------------------------
// attention: per-batch hidden_h . eW[0:512] (+bias) -> base[b]
// ---------------------------------------------------------------------------
__global__ __launch_bounds__(128) void att_base_kernel(const float* __restrict__ hidh,
                                                       const float* __restrict__ eW,
                                                       const float* __restrict__ eb,
                                                       float* __restrict__ base) {
  const int b = blockIdx.x, tid = threadIdx.x;
  __shared__ float red[128];
  float acc = 0.f;
  for (int k = tid; k < HS; k += 128) acc += hidh[(size_t)b * HS + k] * eW[k];
  red[tid] = acc; __syncthreads();
  for (int off = 64; off > 0; off >>= 1) {
    if (tid < off) red[tid] += red[tid + off];
    __syncthreads();
  }
  if (tid == 0) base[b] = red[0] + eb[0];
}

// energies + softmax over S + context (written as bf16 for the decoder GEMM)
__global__ __launch_bounds__(256) void att_main_kernel(const float* __restrict__ outF,
                                                       const float* __restrict__ outB,
                                                       const float* __restrict__ eW,
                                                       const float* __restrict__ base,
                                                       __bf16* __restrict__ ctx_bf) {
  const int b = blockIdx.x, tid = threadIdx.x;
  __shared__ float red[256];
  __shared__ float attn[SLEN];

  for (int s = 0; s < SLEN; ++s) {
    const float* pf = outF + ((size_t)s * BATCH + b) * HS;
    const float* pb = outB + ((size_t)s * BATCH + b) * HS;
    float acc = 0.f;
    for (int k = tid; k < HS; k += 256)
      acc += pf[k] * eW[512 + k] + pb[k] * eW[1024 + k];
    red[tid] = acc; __syncthreads();
    for (int off = 128; off > 0; off >>= 1) {
      if (tid < off) red[tid] += red[tid + off];
      __syncthreads();
    }
    if (tid == 0) attn[s] = fmaxf(base[b] + red[0], 0.f);  // relu
    __syncthreads();
  }
  if (tid == 0) {
    float mx = attn[0];
    for (int s = 1; s < SLEN; ++s) mx = fmaxf(mx, attn[s]);
    float sum = 0.f;
    for (int s = 0; s < SLEN; ++s) { attn[s] = __expf(attn[s] - mx); sum += attn[s]; }
    float inv = 1.f / sum;
    for (int s = 0; s < SLEN; ++s) attn[s] *= inv;
  }
  __syncthreads();
  for (int k = tid; k < 2 * HS; k += 256) {
    const float* srcp = (k < HS) ? (outF + (size_t)b * HS + k)
                                 : (outB + (size_t)b * HS + (k - HS));
    float acc = 0.f;
    for (int s = 0; s < SLEN; ++s) acc += attn[s] * srcp[(size_t)s * BATCH * HS];
    ctx_bf[(size_t)b * (2 * HS) + k] = (__bf16)acc;
  }
}

// ---------------------------------------------------------------------------
// decoder: g[t] = emb_d[t] @ Wih_emb^T + g_base[b] ; LSTM pointwise -> bf16 h
// grid (NTOK*B/64, HS/32), same wave layout as encoder.
// ---------------------------------------------------------------------------
__global__ __launch_bounds__(256) void dec_step_kernel(const __bf16* __restrict__ embd,
                                                       const __bf16* __restrict__ wemb, // dec_wih_bf + 1024, stride 1280
                                                       const float* __restrict__ gbase, // [B][4HS]
                                                       const float* __restrict__ hc,    // hidden_c [B][HS]
                                                       __bf16* __restrict__ hout) {
  const int tid  = threadIdx.x;
  const int wave = tid >> 5, lane = tid & 31;
  const int waveM = wave >> 2, gate = wave & 3;
  const int half = lane >> 4, l16 = lane & 15;
  const int rowBase = blockIdx.x * 64 + waveM * 32;
  const int colGate = blockIdx.y * 32;

  const v8f vz = {0.f, 0.f, 0.f, 0.f, 0.f, 0.f, 0.f, 0.f};
  v8f acc[2][2] = {{vz, vz}, {vz, vz}};

  const __bf16* a0 = embd + (size_t)(rowBase + l16) * ES;
  const __bf16* a1 = embd + (size_t)(rowBase + 16 + l16) * ES;
  const int j0 = gate * HS + colGate + l16;
  const __bf16* b0 = wemb + (size_t)j0 * 1280;
  const __bf16* b1 = wemb + (size_t)(j0 + 16) * 1280;
  mma_2x2(a0, a1, b0, b1, ES, half, acc);

  __shared__ float g_lds[4][64][33];
  for (int mt = 0; mt < 2; ++mt)
    for (int nt = 0; nt < 2; ++nt)
      for (int r = 0; r < 8; ++r)
        g_lds[gate][waveM * 32 + mt * 16 + half * 8 + r][nt * 16 + l16] = acc[mt][nt][r];
  __syncthreads();

  const int col = tid & 31;
  const int rg  = tid >> 5;
  const int n   = blockIdx.y * 32 + col;
  for (int r = 0; r < 8; ++r) {
    const int rl  = rg * 8 + r;
    const int row = blockIdx.x * 64 + rl;          // row = t*B + b
    const int bb  = row & (BATCH - 1);
    const float* gb = gbase + (size_t)bb * (4 * HS);
    float gi = g_lds[0][rl][col] + gb[n];
    float gf = g_lds[1][rl][col] + gb[HS + n];
    float gg = g_lds[2][rl][col] + gb[2 * HS + n];
    float go = g_lds[3][rl][col] + gb[3 * HS + n];
    float cold = hc[(size_t)bb * HS + n];
    float cn = sigm(gf) * cold + sigm(gi) * tanhf(gg);
    float hn = sigm(go) * tanhf(cn);
    hout[(size_t)row * HS + n] = (__bf16)hn;
  }
}

// ---------------------------------------------------------------------------
// logits: (NTOK*B x HS) @ fc_out^T (67 cols, padded to 5 N-tiles), written
// into d_out with the +1 timestep shift. grid (NTOK*B/128), wave = 16 rows.
// ---------------------------------------------------------------------------
__global__ __launch_bounds__(256) void logits_kernel(const __bf16* __restrict__ A,
                                                     const __bf16* __restrict__ W,
                                                     const float* __restrict__ bias,
                                                     float* __restrict__ out) {
  const int tid  = threadIdx.x;
  const int wave = tid >> 5, lane = tid & 31;
  const int half = lane >> 4, l16 = lane & 15;
  const int rowBase = blockIdx.x * 128 + wave * 16;

  const v8f vz = {0.f, 0.f, 0.f, 0.f, 0.f, 0.f, 0.f, 0.f};
  v8f acc[5] = {vz, vz, vz, vz, vz};

  const __bf16* arow = A + (size_t)(rowBase + l16) * HS;
  const __bf16* brow[5];
  for (int nt = 0; nt < 5; ++nt) {
    int c = nt * 16 + l16;
    if (c > OUTD - 1) c = OUTD - 1;                // clamp pad lanes
    brow[nt] = W + (size_t)c * HS;
  }
  for (int kb = 0; kb < HS; kb += 32) {
    v16bf fa = ld_fragA(arow, kb, half);
    for (int nt = 0; nt < 5; ++nt) {
      v16bf fb = ld_fragB(brow[nt], kb, half);
      acc[nt] = wmma_bf(fa, fb, acc[nt]);
    }
  }
  for (int nt = 0; nt < 5; ++nt) {
    const int c = nt * 16 + l16;
    if (c >= OUTD) continue;
    for (int r = 0; r < 8; ++r) {
      const int row = rowBase + half * 8 + r;      // row = t*B + b
      const int tt  = row >> 9;
      const int bb  = row & (BATCH - 1);
      out[((size_t)(tt + 1) * BATCH + bb) * OUTD + c] = acc[nt][r] + bias[c];
    }
  }
}

// ---------------------------------------------------------------------------
// host side
// ---------------------------------------------------------------------------
extern "C" void kernel_launch(void* const* d_in, const int* in_sizes, int n_in,
                              void* d_out, int out_size, void* d_ws, size_t ws_size,
                              hipStream_t stream) {
  const int*   src     = (const int*)d_in[0];
  const int*   trg     = (const int*)d_in[1];
  const float* enc_emb = (const float*)d_in[2];
  const float* eWihF   = (const float*)d_in[3];
  const float* eWhhF   = (const float*)d_in[4];
  const float* eBihF   = (const float*)d_in[5];
  const float* eBhhF   = (const float*)d_in[6];
  const float* eWihB   = (const float*)d_in[7];
  const float* eWhhB   = (const float*)d_in[8];
  const float* eBihB   = (const float*)d_in[9];
  const float* eBhhB   = (const float*)d_in[10];
  const float* fcHW    = (const float*)d_in[11];
  const float* fcHb    = (const float*)d_in[12];
  const float* fcCW    = (const float*)d_in[13];
  const float* fcCb    = (const float*)d_in[14];
  const float* dec_emb = (const float*)d_in[15];
  const float* dWih    = (const float*)d_in[16];
  const float* dWhh    = (const float*)d_in[17];
  const float* dBih    = (const float*)d_in[18];
  const float* dBhh    = (const float*)d_in[19];
  const float* enW     = (const float*)d_in[20];
  const float* enb     = (const float*)d_in[21];
  const float* foW     = (const float*)d_in[22];
  const float* fob     = (const float*)d_in[23];
  float* out = (float*)d_out;

  char* w = (char*)d_ws;
  size_t off = 0;
  auto alloc = [&](size_t bytes) -> void* {
    void* p = w + off;
    off = (off + bytes + 255) & ~(size_t)255;
    return p;
  };

  // bf16 weight / activation buffers
  __bf16* wihF_bf = (__bf16*)alloc((size_t)2048 * 256 * 2);
  __bf16* whhF_bf = (__bf16*)alloc((size_t)2048 * 512 * 2);
  __bf16* wihB_bf = (__bf16*)alloc((size_t)2048 * 256 * 2);
  __bf16* whhB_bf = (__bf16*)alloc((size_t)2048 * 512 * 2);
  __bf16* dWih_bf = (__bf16*)alloc((size_t)2048 * 1280 * 2);
  __bf16* dWhh_bf = (__bf16*)alloc((size_t)2048 * 512 * 2);
  __bf16* fcHW_bf = (__bf16*)alloc((size_t)512 * 1024 * 2);
  __bf16* fcCW_bf = (__bf16*)alloc((size_t)512 * 1024 * 2);
  __bf16* foW_bf  = (__bf16*)alloc((size_t)OUTD * 512 * 2);
  __bf16* emb_bf  = (__bf16*)alloc((size_t)SLEN * BATCH * ES * 2);
  __bf16* embd_bf = (__bf16*)alloc((size_t)NTOK * BATCH * ES * 2);
  __bf16* hbF[2]  = {(__bf16*)alloc((size_t)BATCH * HS * 2),
                     (__bf16*)alloc((size_t)BATCH * HS * 2)};
  __bf16* hbB[2]  = {(__bf16*)alloc((size_t)BATCH * HS * 2),
                     (__bf16*)alloc((size_t)BATCH * HS * 2)};
  __bf16* cf_bf   = (__bf16*)alloc((size_t)BATCH * HS * 2);
  __bf16* cb_bf   = (__bf16*)alloc((size_t)BATCH * HS * 2);
  __bf16* hidh_bf = (__bf16*)alloc((size_t)BATCH * HS * 2);
  __bf16* ctx_bf  = (__bf16*)alloc((size_t)BATCH * 2 * HS * 2);
  __bf16* hdec_bf = (__bf16*)alloc((size_t)NTOK * BATCH * HS * 2);
  // fp32 buffers
  float* c_f    = (float*)alloc((size_t)BATCH * HS * 4);
  float* c_b    = (float*)alloc((size_t)BATCH * HS * 4);
  float* out_f  = (float*)alloc((size_t)SLEN * BATCH * HS * 4);
  float* out_b  = (float*)alloc((size_t)SLEN * BATCH * HS * 4);
  float* hid_h  = (float*)alloc((size_t)BATCH * HS * 4);
  float* hid_c  = (float*)alloc((size_t)BATCH * HS * 4);
  float* basebf = (float*)alloc((size_t)BATCH * 4);
  float* g_base = (float*)alloc((size_t)BATCH * 4 * HS * 4);

  auto cvt = [&](const float* in, __bf16* o, int n) {
    cvt_kernel<<<(n + 255) / 256, 256, 0, stream>>>(in, o, n);
  };
  cvt(eWihF, wihF_bf, 2048 * 256);
  cvt(eWhhF, whhF_bf, 2048 * 512);
  cvt(eWihB, wihB_bf, 2048 * 256);
  cvt(eWhhB, whhB_bf, 2048 * 512);
  cvt(dWih, dWih_bf, 2048 * 1280);
  cvt(dWhh, dWhh_bf, 2048 * 512);
  cvt(fcHW, fcHW_bf, 512 * 1024);
  cvt(fcCW, fcCW_bf, 512 * 1024);
  cvt(foW, foW_bf, OUTD * 512);

  gather_kernel<<<(SLEN * BATCH * ES + 255) / 256, 256, 0, stream>>>(src, enc_emb, emb_bf, SLEN * BATCH);
  gather_kernel<<<(NTOK * BATCH * ES + 255) / 256, 256, 0, stream>>>(trg, dec_emb, embd_bf, NTOK * BATCH);

  // encoder recurrence (both directions per launch)
  EncParams ep{};
  ep.emb = emb_bf;
  ep.wih0 = wihF_bf; ep.whh0 = whhF_bf; ep.wih1 = wihB_bf; ep.whh1 = whhB_bf;
  ep.bih0 = eBihF; ep.bhh0 = eBhhF; ep.bih1 = eBihB; ep.bhh1 = eBhhB;
  ep.c0 = c_f; ep.c1 = c_b; ep.out0 = out_f; ep.out1 = out_b;
  for (int t = 0; t < SLEN; ++t) {
    ep.t = t;
    ep.hrd0 = hbF[t & 1];       ep.hrd1 = hbB[t & 1];
    ep.hwr0 = hbF[(t + 1) & 1]; ep.hwr1 = hbB[(t + 1) & 1];
    enc_step_kernel<<<dim3(BATCH / 64, HS / 32, 2), 256, 0, stream>>>(ep);
  }
  // final bf16 h lives in buffer index (SLEN & 1) == 0
  cvt(c_f, cf_bf, BATCH * HS);
  cvt(c_b, cb_bf, BATCH * HS);

  // hidden_h = [hf|hb] @ fc_h_W^T + b
  GemmParams gh{};
  gh.A0 = hbF[0]; gh.lda0 = HS; gh.K0 = HS;
  gh.A1 = hbB[0]; gh.lda1 = HS; gh.K1 = HS;
  gh.B0 = fcHW_bf; gh.ldb0 = 1024;
  gh.B1 = fcHW_bf + 512; gh.ldb1 = 1024;
  gh.bias0 = fcHb; gh.bias1 = nullptr;
  gh.Cf = hid_h; gh.Cb = hidh_bf; gh.ldc = HS;
  gemm2seg_kernel<<<dim3(BATCH / 64, HS / 128), 256, 0, stream>>>(gh);

  // hidden_c = [cf|cb] @ fc_c_W^T + b
  GemmParams gc = gh;
  gc.A0 = cf_bf; gc.A1 = cb_bf;
  gc.B0 = fcCW_bf; gc.B1 = fcCW_bf + 512;
  gc.bias0 = fcCb;
  gc.Cf = hid_c; gc.Cb = nullptr;
  gemm2seg_kernel<<<dim3(BATCH / 64, HS / 128), 256, 0, stream>>>(gc);

  // attention
  att_base_kernel<<<BATCH, 128, 0, stream>>>(hid_h, enW, enb, basebf);
  att_main_kernel<<<BATCH, 256, 0, stream>>>(out_f, out_b, enW, basebf, ctx_bf);

  // decoder shared term: g_base = ctx @ Wih[:, :1024]^T + hidden_h @ Whh^T + biases
  GemmParams gb{};
  gb.A0 = ctx_bf;  gb.lda0 = 2 * HS; gb.K0 = 2 * HS;
  gb.A1 = hidh_bf; gb.lda1 = HS;     gb.K1 = HS;
  gb.B0 = dWih_bf; gb.ldb0 = 1280;
  gb.B1 = dWhh_bf; gb.ldb1 = 512;
  gb.bias0 = dBih; gb.bias1 = dBhh;
  gb.Cf = g_base; gb.Cb = nullptr; gb.ldc = 4 * HS;
  gemm2seg_kernel<<<dim3(BATCH / 64, (4 * HS) / 128), 256, 0, stream>>>(gb);

  // decoder per-token GEMM + LSTM pointwise
  dec_step_kernel<<<dim3((NTOK * BATCH) / 64, HS / 32), 256, 0, stream>>>(
      embd_bf, dWih_bf + 1024, g_base, hid_c, hdec_bf);

  // logits into d_out (shifted by one timestep) + zero first slice
  logits_kernel<<<(NTOK * BATCH) / 128, 256, 0, stream>>>(hdec_bf, foW_bf, fob, out);
  zero_kernel<<<(BATCH * OUTD + 255) / 256, 256, 0, stream>>>(out, BATCH * OUTD);
}
